// NMS_2860448219381
// MI455X (gfx1250) — compile-verified, matrix operand order checked by reading, not run on hardware
//
#include <hip/hip_runtime.h>

#define CONF_THRES 0.25f
#define IOU_THRES  0.45f
#define NCLS    80
#define NSTR    85          // floats per prediction row
#define KPRE    1024        // padded top-K (reference K_PRE = 1000)
#define KSEL    1000
#define MAXDET  300
#define CAND    2048        // candidate pool for exact top-K sort

typedef __attribute__((ext_vector_type(16))) _Float16 v16h;
typedef __attribute__((ext_vector_type(8)))  float    v8f;

// ---------------------------------------------------------------------------
// Kernel 1: decode xywh->xyxy, conf = obj * max_c(cls), argmax class.
// LDS-staged so global reads are fully coalesced (85-float rows are awful
// for per-thread direct loads). 137 MB streaming pass -> ~6 us at 23.3 TB/s.
// ---------------------------------------------------------------------------
extern "C" __global__ void k_decode(const float* __restrict__ pred,
                                    float* __restrict__ boxes,
                                    int*   __restrict__ clsb,
                                    float* __restrict__ scoreb,
                                    int N) {
  extern __shared__ float dynsm[];           // 256 * 85 floats = 87,040 B
  float* tile = dynsm;
  const int b  = blockIdx.y;
  const int n0 = blockIdx.x * 256;
  int cnt = N - n0; if (cnt > 256) cnt = 256;
  const int tot = cnt * NSTR;
  const float* src = pred + ((size_t)b * N + n0) * NSTR;
  for (int k = threadIdx.x; k < tot; k += 256) tile[k] = src[k];
  __syncthreads();

  const int t = threadIdx.x;
  if (t < cnt) {
    const float* p = &tile[t * NSTR];        // stride 85 % 64 banks -> conflict free
    const float obj = p[4];
    float best = p[5]; int bi = 0;
#pragma unroll 4
    for (int c = 1; c < NCLS; ++c) {
      float v = p[5 + c];
      if (v > best) { best = v; bi = c; }    // strict > == first-argmax (jnp)
    }
    const float conf = best * obj;
    const float cx = p[0], cy = p[1], w = p[2], h = p[3];
    const size_t n = (size_t)b * N + n0 + t;
    float4 bb = make_float4(cx - 0.5f * w, cy - 0.5f * h,
                            cx + 0.5f * w, cy + 0.5f * h);
    *(float4*)&boxes[n * 4] = bb;
    clsb[n]   = bi;
    scoreb[n] = conf;
  }
}

// ---------------------------------------------------------------------------
// Kernel 2: exact top-1000 per image. One 1024-thread workgroup per image.
//  a) LDS histogram of scores above CONF_THRES (1024 bins)
//  b) threshold bin so that #above >= 1000
//  c) deterministic (index-ordered) ballot-scan compaction into <=2048 pool
//  d) LDS bitonic sort (score desc, index asc tie-break), emit first 1000
// ---------------------------------------------------------------------------
extern "C" __global__ void __launch_bounds__(1024)
k_topk(const float* __restrict__ scoreb,
       unsigned*    __restrict__ sel_idx,
       float*       __restrict__ sel_score,
       int N) {
  __shared__ unsigned hist[1024];
  __shared__ float    ckey[CAND];
  __shared__ int      cidx[CAND];
  __shared__ unsigned wscr[33];
  __shared__ int      s_tbin;

  const int b = blockIdx.x;
  const int t = threadIdx.x;
  const unsigned lane = t & 31, wid = t >> 5;
  const float* sc = scoreb + (size_t)b * N;

  hist[t] = 0;
  __syncthreads();
  for (int i = t; i < N; i += 1024) {
    __builtin_prefetch(sc + i + 4096, 0, 1);     // global_prefetch_b8
    const float s = sc[i];
    if (s > CONF_THRES) {
      int bin = (int)((s - CONF_THRES) * (1024.0f / 0.75f));
      bin = bin < 0 ? 0 : (bin > 1023 ? 1023 : bin);
      atomicAdd(&hist[bin], 1u);
    }
  }
  __syncthreads();
  if (t == 0) {
    int run = 0, tb = 0;
    for (int x = 1023; x >= 0; --x) {
      run += (int)hist[x];
      if (run >= KSEL) { tb = x; break; }
    }
    s_tbin = tb;                                  // if <1000 valid: tb==0, take all
  }
  __syncthreads();
  const int tbin = s_tbin;

  unsigned base = 0;
  const int nchunk = (N + 1023) / 1024;
  for (int pass = 0; pass < 2; ++pass) {          // 0: bin>t, 1: bin==t
    for (int c = 0; c < nchunk; ++c) {
      const int i = c * 1024 + t;
      const float s = (i < N) ? sc[i] : -1.0f;
      bool p = false;
      if (s > CONF_THRES) {
        int bin = (int)((s - CONF_THRES) * (1024.0f / 0.75f));
        bin = bin < 0 ? 0 : (bin > 1023 ? 1023 : bin);
        p = (pass == 0) ? (bin > tbin) : (bin == tbin);
      }
      __syncthreads();                            // protect wscr reuse
      const unsigned m = __builtin_amdgcn_ballot_w32(p);
      if (lane == 0) wscr[wid] = __popc(m);
      __syncthreads();
      if (t == 0) {
        unsigned r = 0;
        for (int w2 = 0; w2 < 32; ++w2) { unsigned x = wscr[w2]; wscr[w2] = r; r += x; }
        wscr[32] = r;
      }
      __syncthreads();
      const unsigned rank = base + wscr[wid] + __popc(m & ((1u << lane) - 1u));
      if (p && rank < CAND) { ckey[rank] = s; cidx[rank] = i; }
      base += wscr[32];                           // uniform across block
    }
  }
  const unsigned filled = base < CAND ? base : CAND;
  __syncthreads();
  for (int k = t; k < CAND; k += 1024)
    if ((unsigned)k >= filled) { ckey[k] = -2.0f; cidx[k] = 0x7fffffff; }
  __syncthreads();

  // bitonic sort, ranking order = (score desc, index asc)
  for (int ksz = 2; ksz <= CAND; ksz <<= 1) {
    for (int j = ksz >> 1; j > 0; j >>= 1) {
      for (int u = t; u < CAND; u += 1024) {
        const int v = u ^ j;
        if (v > u) {
          const float ka = ckey[u], kb = ckey[v];
          const int   ia = cidx[u], ib = cidx[v];
          const bool before = (ka > kb) || (ka == kb && ia < ib);
          const bool doswap = ((u & ksz) == 0) ? !before : before;
          if (doswap) { ckey[u] = kb; ckey[v] = ka; cidx[u] = ib; cidx[v] = ia; }
        }
      }
      __syncthreads();
    }
  }

  if (t < KPRE) {
    const bool real = (t < KSEL);
    const float s = real ? ckey[t] : -2.0f;
    int ii = real ? cidx[t] : 0;
    if (ii == 0x7fffffff) ii = 0;                 // padded slot, guarded by score
    sel_idx[b * KPRE + t]   = (unsigned)ii;
    sel_score[b * KPRE + t] = s;
  }
}

// ---------------------------------------------------------------------------
// Kernel 3: per-image IoU bitmask + greedy suppression + emit top-300.
// One 1024-thread workgroup per image; everything lives in ~160 KB LDS
// (CDNA5: up to 320 KB per workgroup -> 2 WGs per WGP).
//   * 16x16 tiles per wave; pairwise (area_i + area_j) computed on the XDL
//     pipe via v_wmma_f32_16x16x32_f16 (A=[area_i,1], B=[1;area_j]) while the
//     VALU does the min/max intersection (co-executable).
//   * class-offset (c*MAX_WH) replaced exactly by cls_i==cls_j test.
//   * ballot_w32 packs two 16-bit bitmask rows per 16x16 tile.
//   * greedy scan: single wave, 64 u16 keep-words, 1000 iterations in LDS.
// ---------------------------------------------------------------------------
extern "C" __global__ void __launch_bounds__(1024)
k_nms(const float*    __restrict__ boxes,
      const int*      __restrict__ clsb,
      const unsigned* __restrict__ sel_idx,
      const float*    __restrict__ sel_score,
      float*          __restrict__ out,
      int N) {
  extern __shared__ float dynsm[];
  float* bx1 = dynsm;                 // [KPRE]
  float* by1 = bx1 + KPRE;
  float* bx2 = by1 + KPRE;
  float* by2 = bx2 + KPRE;
  float* bar = by2 + KPRE;
  float* bsc = bar + KPRE;
  int*   bcl = (int*)(bsc + KPRE);
  unsigned short* mask  = (unsigned short*)(bcl + KPRE);   // [KPRE][64] = 128 KB
  unsigned short* keepw = mask + KPRE * 64;                // [64]
  unsigned*       pre   = (unsigned*)(keepw + 64);         // [64]

  const int b = blockIdx.x;
  const int t = threadIdx.x;
  const int wid = t >> 5, lane = t & 31;
  const int hi = lane >> 4, jj = lane & 15;

  // zero output slice (reference zero-fills unused detection rows)
  for (int k = t; k < MAXDET * 6; k += 1024) out[(size_t)b * MAXDET * 6 + k] = 0.0f;
  // zero bitmask (lower-triangle tiles are skipped and must read as 0)
  for (int k = t; k < KPRE * 64; k += 1024) mask[k] = 0;

  // gather candidate boxes into LDS (SoA)
  {
    const float s = sel_score[b * KPRE + t];
    const bool valid = s > CONF_THRES;
    float x1 = 0, y1 = 0, x2 = 0, y2 = 0; int cl = -1;
    if (valid) {
      const unsigned idx = sel_idx[b * KPRE + t];
      const float* bp = &boxes[((size_t)b * N + idx) * 4];
      x1 = bp[0]; y1 = bp[1]; x2 = bp[2]; y2 = bp[3];
      cl = clsb[(size_t)b * N + idx];
    }
    bx1[t] = x1; by1[t] = y1; bx2[t] = x2; by2[t] = y2;
    bar[t] = (x2 - x1) * (y2 - y1);
    bsc[t] = s; bcl[t] = cl;
  }
  __syncthreads();

  // ---- suppression bitmask: 64x64 grid of 16x16 tiles, upper triangle ----
  for (int tile = wid; tile < 64 * 64; tile += 32) {       // wave-uniform
    const int ti = tile >> 6, tj = tile & 63;
    if (tj < ti) continue;                                 // j>i only
    const int i0 = ti << 4, j0 = tj << 4;
    const int j = j0 + jj;
    const float jx1 = bx1[j], jy1 = by1[j], jx2 = bx2[j], jy2 = by2[j];
    const int jc = bcl[j];

    // XDL: D[M][n] = area_i[i0+M]*1 + 1*area_j[j0+n]  (rank-2 outer sum)
    v16h a = {}, bv = {};
    const int M = lane & 15;
    if (hi == 0) {
      a[0]  = (_Float16)bar[i0 + M];   // A: K=0 column = area_i
      a[1]  = (_Float16)1.0f;          // A: K=1 column = 1
      bv[0] = (_Float16)1.0f;          // B: K=0 row   = 1
      bv[1] = (_Float16)bar[j0 + M];   // B: K=1 row   = area_j (n == M here)
    }
    v8f c = {};
    v8f d = __builtin_amdgcn_wmma_f32_16x16x32_f16(
        false, a, false, bv, (short)0, c, false, false);

#pragma unroll
    for (int r = 0; r < 8; ++r) {
      const int i = i0 + r + (hi << 3);
      const float w = fminf(bx2[i], jx2) - fmaxf(bx1[i], jx1);
      const float h = fminf(by2[i], jy2) - fmaxf(by1[i], jy1);
      const float inter = fmaxf(w, 0.0f) * fmaxf(h, 0.0f);
      const float denom = d[r] - inter + 1e-9f;            // areaI+areaJ-inter
      const bool bit = (bcl[i] == jc) && (j > i) && (inter > IOU_THRES * denom);
      const unsigned bal = __builtin_amdgcn_ballot_w32(bit);
      if (lane == 0) {
        mask[(i0 + r) * 64 + tj]     = (unsigned short)(bal & 0xFFFFu);  // rows i0+r
        mask[(i0 + r + 8) * 64 + tj] = (unsigned short)(bal >> 16);      // rows i0+r+8
      }
    }
  }
  __syncthreads();

  // init keep mask from validity
  if (t < 64) {
    unsigned short w = 0;
    for (int k2 = 0; k2 < 16; ++k2) {
      const int e = t * 16 + k2;
      if (e < KSEL && bsc[e] > CONF_THRES) w |= (unsigned short)(1u << k2);
    }
    keepw[t] = w;
  }
  __syncthreads();

  // greedy scan: one wave, each lane owns keep-words {lane, lane+32}
  if (wid == 0) {
    volatile unsigned short* kp = keepw;
    for (int i = 0; i < KSEL; ++i) {
      const unsigned wbits = kp[i >> 4];
      if ((wbits >> (i & 15)) & 1u) {
        kp[lane]      = (unsigned short)(kp[lane]      & (unsigned short)~mask[i * 64 + lane]);
        kp[lane + 32] = (unsigned short)(kp[lane + 32] & (unsigned short)~mask[i * 64 + lane + 32]);
      }
    }
  }
  __syncthreads();

  // rank (cumsum of keep) and emit first 300 detections
  if (t == 0) {
    unsigned run = 0;
    for (int w2 = 0; w2 < 64; ++w2) { pre[w2] = run; run += __popc((unsigned)keepw[w2]); }
  }
  __syncthreads();
  if (t < KSEL) {
    const unsigned wv = keepw[t >> 4];
    if ((wv >> (t & 15)) & 1u) {
      const unsigned rank = pre[t >> 4] + __popc(wv & ((1u << (t & 15)) - 1u));
      if (rank < MAXDET) {
        float* o = &out[((size_t)b * MAXDET + rank) * 6];
        o[0] = bx1[t]; o[1] = by1[t]; o[2] = bx2[t]; o[3] = by2[t];
        o[4] = bsc[t]; o[5] = (float)bcl[t];
      }
    }
  }
}

// ---------------------------------------------------------------------------
extern "C" void kernel_launch(void* const* d_in, const int* in_sizes, int n_in,
                              void* d_out, int out_size, void* d_ws, size_t ws_size,
                              hipStream_t stream) {
  const float* pred = (const float*)d_in[0];
  const int B = 16;
  const int N = in_sizes[0] / (B * NSTR);       // 25200

  // workspace layout (all 16B-aligned slabs)
  float*    boxes     = (float*)d_ws;                               // B*N*4 f32
  int*      clsb      = (int*)(boxes + (size_t)B * N * 4);          // B*N i32
  float*    scoreb    = (float*)(clsb + (size_t)B * N);             // B*N f32
  unsigned* sel_idx   = (unsigned*)(scoreb + (size_t)B * N);        // B*KPRE u32
  float*    sel_score = (float*)(sel_idx + (size_t)B * KPRE);       // B*KPRE f32

  dim3 g1((N + 255) / 256, B);
  k_decode<<<g1, 256, 256 * NSTR * sizeof(float), stream>>>(pred, boxes, clsb, scoreb, N);

  k_topk<<<B, 1024, 0, stream>>>(scoreb, sel_idx, sel_score, N);

  const size_t smem3 = (size_t)KPRE * 6 * 4     // box SoA + area + score
                     + (size_t)KPRE * 4         // cls
                     + (size_t)KPRE * 64 * 2    // suppression bitmask (128 KB)
                     + 64 * 2                   // keep words
                     + 64 * 4;                  // prefix
  k_nms<<<B, 1024, smem3, stream>>>(boxes, clsb, sel_idx, sel_score, (float*)d_out, N);
}
